// AttentionWithDynamicOutput_22170621182631
// MI455X (gfx1250) — compile-verified, matrix operand order checked by reading, not run on hardware
//
#include <hip/hip_runtime.h>
#include <hip/hip_bf16.h>

// ---------------------------------------------------------------------------
// CDNA5 (gfx1250, wave32) WMMA attention + token pruning pipeline.
// Matmuls: v_wmma_f32_16x16x32_bf16.  LDS staging: global_load_async_to_lds
// (ASYNCcnt-tracked, double-buffered) when the toolchain exposes it.
// ---------------------------------------------------------------------------

typedef __attribute__((ext_vector_type(16))) __bf16 v16bf;
typedef __attribute__((ext_vector_type(8)))  float  v8f;

#ifndef __has_builtin
#define __has_builtin(x) 0
#endif

#if __has_builtin(__builtin_amdgcn_global_load_async_to_lds_b128)
#define ASYNC_LDS 1
// Builtin signature (from hipcc diagnostic): param0 = global (AS1) pointer to
// GCC-vector int4, param1 = LDS (AS3) pointer to int4, then imm offset, imm cpol.
typedef int gcc_int4 __attribute__((vector_size(16)));
typedef __attribute__((address_space(1))) gcc_int4 as1_int4;
typedef __attribute__((address_space(3))) gcc_int4 as3_int4;
#else
#define ASYNC_LDS 0
#endif

// 16-byte global -> LDS copy.  Async path bypasses VGPRs and is tracked by
// ASYNCcnt; fallback is a plain load/store pair.
__device__ __forceinline__ void async_copy16(const void* g, void* l) {
#if ASYNC_LDS
    __builtin_amdgcn_global_load_async_to_lds_b128((as1_int4*)g, (as3_int4*)l,
                                                   0, 0);
#else
    *(uint4*)l = *(const uint4*)g;
#endif
}

__device__ __forceinline__ void async_wait_all() {
#if ASYNC_LDS
#if __has_builtin(__builtin_amdgcn_s_wait_asynccnt)
    __builtin_amdgcn_s_wait_asynccnt(0);
#else
    asm volatile("s_wait_asynccnt 0x0" ::: "memory");
#endif
#endif
}

__device__ __forceinline__ unsigned short f2bf(float f) {
    unsigned int x = __float_as_uint(f);
    unsigned int r = x + 0x7FFFu + ((x >> 16) & 1u);   // round-to-nearest-even
    return (unsigned short)(r >> 16);
}

union FragB16 {
    v16bf v;
    unsigned short h[16];
    unsigned int   u[8];
};

// K index covered by VGPR j within a 16-bit A/B fragment (ISA 7.12.2):
// lanes 0-15 / 16-31 split the K range into +0 / +8 halves, VGPRs 0-3 hold
// K 0..7 (pairs), VGPRs 4-7 hold K 16..23 (pairs).
__device__ __forceinline__ int frag_kbase(int j, int half) {
    return ((j < 4) ? (2 * j) : (16 + 2 * (j - 4))) + 8 * half;
}

// Fragment whose "row" dimension (M for A, N for Kᵀ-as-B) is the slow LDS
// axis and K is contiguous: pairs load as one 32-bit LDS read.
__device__ __forceinline__ v16bf frag_rowmajor(const unsigned short* p, int ld,
                                               int row0, int k0, int lane) {
    int half = lane >> 4, r = lane & 15;
    FragB16 f;
    const unsigned short* rowp = p + (row0 + r) * ld + k0;
#pragma unroll
    for (int j = 0; j < 8; ++j) {
        int kb = frag_kbase(j, half);
        f.u[j] = *(const unsigned int*)(rowp + kb);
    }
    return f.v;
}

// B fragment from a K-major [K][N] LDS tile.
__device__ __forceinline__ v16bf frag_kmajor(const unsigned short* p, int ld,
                                             int k0, int n0, int lane) {
    int half = lane >> 4, n = lane & 15;
    FragB16 f;
#pragma unroll
    for (int j = 0; j < 8; ++j) {
        int kb = frag_kbase(j, half) + k0;
        f.h[2 * j]     = p[(size_t)kb * ld + n0 + n];
        f.h[2 * j + 1] = p[(size_t)(kb + 1) * ld + n0 + n];
    }
    return f.v;
}

// A fragment from an f32 LDS tile, converting to bf16 on the fly.
__device__ __forceinline__ v16bf frag_rowmajor_f32(const float* p, int ld,
                                                   int row0, int k0, int lane) {
    int half = lane >> 4, r = lane & 15;
    FragB16 f;
    const float* rowp = p + (row0 + r) * ld + k0;
#pragma unroll
    for (int j = 0; j < 8; ++j) {
        int kb = frag_kbase(j, half);
        f.h[2 * j]     = f2bf(rowp[kb]);
        f.h[2 * j + 1] = f2bf(rowp[kb + 1]);
    }
    return f.v;
}

__device__ __forceinline__ v8f wmma_bf16(v16bf a, v16bf b, v8f c) {
    return __builtin_amdgcn_wmma_f32_16x16x32_bf16(
        /*neg_a=*/false, a, /*neg_b=*/false, b,
        /*c_mod=*/(short)0, c, /*reuse_a=*/false, /*reuse_b=*/false);
}

// ---------------------------------------------------------------------------
// f32 -> bf16 conversion
// ---------------------------------------------------------------------------
__global__ void f32_to_bf16_kernel(const float* __restrict__ in,
                                   unsigned short* __restrict__ out, int n) {
    int i = blockIdx.x * blockDim.x + threadIdx.x;
    int stride = gridDim.x * blockDim.x;
    for (; i < n; i += stride) out[i] = f2bf(in[i]);
}

// ---------------------------------------------------------------------------
// GEMM: C[M,N] = A[M,K](bf16) * B[K,N](bf16) + bias.
// BM=64, BN=256, BK=32; 8 waves as 2x4, each wave a 32x64 block of 16x16
// WMMA tiles (8 wmma : 6 fragment loads per K-step).  LDS double-buffered,
// staged with async-to-LDS copies overlapping WMMA compute.
// M % 64 == 0, N % 256 == 0, K % 32 == 0 (true for all uses here).
// ---------------------------------------------------------------------------
template <bool OUT_BF16>
__global__ __launch_bounds__(256)
void gemm_bf16_kernel(const unsigned short* __restrict__ A,
                      const unsigned short* __restrict__ B,
                      const float* __restrict__ bias,
                      void* __restrict__ Cout,
                      int M, int N, int K) {
    constexpr int BM = 64, BN = 256, BK = 32;
    constexpr int LDB = BN + 8;                    // 264: 528B rows, 16B-aligned
    __shared__ unsigned short As[2][BM * BK];      // 2 x 4 KB
    __shared__ unsigned short Bs[2][BK * LDB];     // 2 x 16.5 KB

    const int tid  = threadIdx.x;
    const int lane = tid & 31;
    const int wid  = tid >> 5;
    const int m0 = blockIdx.x * BM;
    const int n0 = blockIdx.y * BN;
    const int waveM = (wid >> 2) * 32;             // 2 wave-rows
    const int waveN = (wid & 3) * 64;              // 4 wave-cols

    const int arow = tid >> 2, acol = (tid & 3) * 8;   // A: 256 x 16B

    v8f acc[2][4];
#pragma unroll
    for (int mi = 0; mi < 2; ++mi)
#pragma unroll
        for (int ni = 0; ni < 4; ++ni)
            acc[mi][ni] = (v8f){0.f, 0.f, 0.f, 0.f, 0.f, 0.f, 0.f, 0.f};

    const int nk = K / BK;

    // stage tile k0 into buffer `buf`
    auto stage = [&](int buf, int k0) {
        async_copy16(&A[(size_t)(m0 + arow) * K + k0 + acol],
                     &As[buf][arow * BK + acol]);
#pragma unroll
        for (int t = 0; t < 4; ++t) {              // B: 1024 x 16B
            int vid = tid + t * 256;
            int row = vid >> 5;
            int col = (vid & 31) * 8;
            async_copy16(&B[(size_t)(k0 + row) * N + n0 + col],
                         &Bs[buf][row * LDB + col]);
        }
    };

    stage(0, 0);
    for (int it = 0; it < nk; ++it) {
        const int cb = it & 1;
        async_wait_all();                           // own async copies done
        __syncthreads();                            // everyone's copies visible
        if (it + 1 < nk) stage(cb ^ 1, (it + 1) * BK);

        v16bf af[2], bfr[4];
#pragma unroll
        for (int mi = 0; mi < 2; ++mi)
            af[mi] = frag_rowmajor(&As[cb][0], BK, waveM + mi * 16, 0, lane);
#pragma unroll
        for (int ni = 0; ni < 4; ++ni)
            bfr[ni] = frag_kmajor(&Bs[cb][0], LDB, 0, waveN + ni * 16, lane);
#pragma unroll
        for (int mi = 0; mi < 2; ++mi)
#pragma unroll
            for (int ni = 0; ni < 4; ++ni)
                acc[mi][ni] = wmma_bf16(af[mi], bfr[ni], acc[mi][ni]);
    }

    // epilogue: C/D layout — VGPR r holds M = r + 8*(lane>>4), N = lane&15
    const int half = lane >> 4, nn = lane & 15;
#pragma unroll
    for (int mi = 0; mi < 2; ++mi)
#pragma unroll
        for (int ni = 0; ni < 4; ++ni) {
            int gn = n0 + waveN + ni * 16 + nn;
            float bv = bias[gn];
#pragma unroll
            for (int r = 0; r < 8; ++r) {
                int gm = m0 + waveM + mi * 16 + r + 8 * half;
                float val = acc[mi][ni][r] + bv;
                if (OUT_BF16)
                    ((unsigned short*)Cout)[(size_t)gm * N + gn] = f2bf(val);
                else
                    ((float*)Cout)[(size_t)gm * N + gn] = val;
            }
        }
}

// ---------------------------------------------------------------------------
// Attention: one block per (b,h).  K/V (224x64, zero-padded) resident in LDS,
// 7 strips of 32 query rows.  Scores via WMMA, softmax via wave reductions,
// CLS attention row accumulated into cls_attn, P*V via WMMA, bf16 out in
// [b,n,h,d] layout so the proj GEMM reads it row-major.
// ---------------------------------------------------------------------------
#define ATT_NP 224
#define ATT_D  64

__global__ __launch_bounds__(256)
void attention_kernel(const unsigned short* __restrict__ qkv,  // [B*197][2304]
                      float* __restrict__ cls_attn,            // [B][196]
                      unsigned short* __restrict__ attn_out) { // [B*197][768]
    extern __shared__ char smem[];
    unsigned short* Ks = (unsigned short*)smem;            // 224*64 bf16
    unsigned short* Vs = Ks + ATT_NP * ATT_D;              // 224*64 bf16
    unsigned short* Qs = Vs + ATT_NP * ATT_D;              // 32*64  bf16
    float*          Sc = (float*)(Qs + 32 * ATT_D);        // 32*224 f32

    const int tid = threadIdx.x, lane = tid & 31, wid = tid >> 5;
    const int b = blockIdx.x / 12, h = blockIdx.x % 12;
    const size_t rowbase = (size_t)b * 197;
    const float scale = 0.125f;                            // d^-0.5, d=64

    // zero the pad rows (tokens 197..223) of K and V
    for (int e = tid; e < (ATT_NP - 197) * (ATT_D / 8); e += 256) {
        int tok = 197 + e / (ATT_D / 8);
        int c   = (e % (ATT_D / 8)) * 8;
        uint4 z = {0, 0, 0, 0};
        *(uint4*)&Ks[tok * ATT_D + c] = z;
        *(uint4*)&Vs[tok * ATT_D + c] = z;
    }
    // async-stage the 197 valid K and V rows
    for (int vi = tid; vi < 197 * (ATT_D / 8); vi += 256) {
        int tok = vi >> 3;
        int c   = (vi & 7) * 8;
        const unsigned short* rp = qkv + (rowbase + tok) * 2304 + h * 64 + c;
        async_copy16(rp + 768,  &Ks[tok * ATT_D + c]);
        async_copy16(rp + 1536, &Vs[tok * ATT_D + c]);
    }
    async_wait_all();
    __syncthreads();

    for (int s = 0; s < 7; ++s) {
        // Q strip: 32x64 = 256 x 16B (one copy per thread)
        {
            int tok  = tid >> 3;
            int c    = (tid & 7) * 8;
            int grow = s * 32 + tok;
            if (grow < 197) {
                async_copy16(qkv + (rowbase + grow) * 2304 + h * 64 + c,
                             &Qs[tok * ATT_D + c]);
            } else {
                uint4 z = {0, 0, 0, 0};
                *(uint4*)&Qs[tok * ATT_D + c] = z;
            }
        }
        async_wait_all();
        __syncthreads();

        // S = Q Kᵀ * scale : 2x14 tiles of 16x16, K-dim 64 (2 wmma steps)
        {
            int mi = wid & 1;
            for (int nt = wid >> 1; nt < 14; nt += 4) {
                v8f accS = {0.f, 0.f, 0.f, 0.f, 0.f, 0.f, 0.f, 0.f};
#pragma unroll
                for (int ks = 0; ks < 2; ++ks) {
                    v16bf a  = frag_rowmajor(Qs, ATT_D, mi * 16, ks * 32, lane);
                    v16bf bb = frag_rowmajor(Ks, ATT_D, nt * 16, ks * 32, lane);
                    accS = wmma_bf16(a, bb, accS);
                }
                int half = lane >> 4, nn = lane & 15;
#pragma unroll
                for (int r = 0; r < 8; ++r)
                    Sc[(mi * 16 + r + 8 * half) * ATT_NP + nt * 16 + nn] =
                        accS[r] * scale;
            }
        }
        __syncthreads();

        // Softmax: each wave handles 4 rows; pad columns contribute 0
        for (int row = wid; row < 32; row += 8) {
            float* srow = Sc + row * ATT_NP;
            float mx = -3.0e38f;
            for (int c = lane; c < ATT_NP; c += 32) {
                float x = (c < 197) ? srow[c] : -3.0e38f;
                mx = fmaxf(mx, x);
            }
#pragma unroll
            for (int off = 16; off > 0; off >>= 1)
                mx = fmaxf(mx, __shfl_xor(mx, off, 32));
            float sum = 0.f;
            for (int c = lane; c < ATT_NP; c += 32) {
                float e = (c < 197) ? __expf(srow[c] - mx) : 0.f;
                srow[c] = e;
                sum += e;
            }
#pragma unroll
            for (int off = 16; off > 0; off >>= 1)
                sum += __shfl_xor(sum, off, 32);
            float inv = 1.f / sum;
            for (int c = lane; c < ATT_NP; c += 32) srow[c] *= inv;
        }
        __syncthreads();

        // CLS attention: global row 0 lives in strip 0, local row 0
        if (s == 0) {
            for (int c = tid + 1; c < 197; c += 256)
                atomicAdd(&cls_attn[(size_t)b * 196 + (c - 1)],
                          Sc[c] * (1.f / 12.f));
        }

        // O = P V : 2x4 tiles (one per wave), K-dim 224 (7 wmma steps)
        {
            int mi = wid >> 2, ni = wid & 3;
            v8f accO = {0.f, 0.f, 0.f, 0.f, 0.f, 0.f, 0.f, 0.f};
#pragma unroll
            for (int ks = 0; ks < 7; ++ks) {
                v16bf a  = frag_rowmajor_f32(Sc, ATT_NP, mi * 16, ks * 32, lane);
                v16bf bb = frag_kmajor(Vs, ATT_D, ks * 32, ni * 16, lane);
                accO = wmma_bf16(a, bb, accO);
            }
            int half = lane >> 4, nn = lane & 15;
#pragma unroll
            for (int r = 0; r < 8; ++r) {
                int m = mi * 16 + r + 8 * half;
                int grow = s * 32 + m;
                if (grow < 197)
                    attn_out[(rowbase + grow) * 768 + h * 64 + ni * 16 + nn] =
                        f2bf(accO[r]);
            }
        }
        __syncthreads();
    }
}

// ---------------------------------------------------------------------------
// Per-batch top-k (138 of 196) via bitonic sort of 256 (value,index) pairs.
// Also emits the broadcast index tensor.
// ---------------------------------------------------------------------------
__global__ __launch_bounds__(256)
void topk_kernel(const float* __restrict__ cls,
                 float* __restrict__ idx_out,     // [B][138]
                 float* __restrict__ index_out) { // [B][138][768]
    __shared__ float v[256];
    __shared__ int   ix[256];
    int b = blockIdx.x, t = threadIdx.x;
    v[t]  = (t < 196) ? cls[(size_t)b * 196 + t] : -3.0e38f;
    ix[t] = t;
    __syncthreads();
    for (int k = 2; k <= 256; k <<= 1) {
        for (int j = k >> 1; j > 0; j >>= 1) {
            int p = t ^ j;
            if (p > t) {
                bool descend = ((t & k) == 0);
                bool sw = descend ? (v[t] < v[p]) : (v[t] > v[p]);
                if (sw) {
                    float tv = v[t]; v[t] = v[p]; v[p] = tv;
                    int   ti = ix[t]; ix[t] = ix[p]; ix[p] = ti;
                }
            }
            __syncthreads();
        }
    }
    if (t < 138) idx_out[(size_t)b * 138 + t] = (float)ix[t];
    __syncthreads();
    for (int e = t; e < 138 * 768; e += 256) {
        int j = e / 768;
        index_out[(size_t)b * 138 * 768 + e] = (float)ix[j];
    }
}

__global__ void init_kernel(float* __restrict__ cls, float* __restrict__ left,
                            int n) {
    int i = blockIdx.x * blockDim.x + threadIdx.x;
    if (i < n) cls[i] = 0.f;
    if (i == 0) *left = 138.f;
}

// ---------------------------------------------------------------------------
// Launch
// ---------------------------------------------------------------------------
extern "C" void kernel_launch(void* const* d_in, const int* in_sizes, int n_in,
                              void* d_out, int out_size, void* d_ws,
                              size_t ws_size, hipStream_t stream) {
    const float* x      = (const float*)d_in[0];
    const float* qkv_w  = (const float*)d_in[1];
    const float* qkv_b  = (const float*)d_in[2];
    const float* proj_w = (const float*)d_in[3];
    const float* proj_b = (const float*)d_in[4];

    constexpr int B = 64, N = 197, C = 768, H = 12, LEFT = 138;
    constexpr int M = B * N;  // 12608

    float* out_x     = (float*)d_out;                       // M*C
    float* out_index = out_x + (size_t)M * C;               // B*LEFT*C
    float* out_idx   = out_index + (size_t)B * LEFT * C;    // B*LEFT
    float* out_cls   = out_idx + (size_t)B * LEFT;          // B*(N-1)
    float* out_left  = out_cls + (size_t)B * (N - 1);       // 1

    // Workspace partition (bf16 buffers, 256B-aligned): ~97 MB total
    char* ws = (char*)d_ws;
    auto alloc = [&](size_t bytes) {
        char* p = ws;
        ws += (bytes + 255) & ~(size_t)255;
        return p;
    };
    unsigned short* xb    = (unsigned short*)alloc((size_t)M * C * 2);
    unsigned short* wqkv  = (unsigned short*)alloc((size_t)C * 3 * C * 2);
    unsigned short* wproj = (unsigned short*)alloc((size_t)C * C * 2);
    unsigned short* qkvo  = (unsigned short*)alloc((size_t)M * 3 * C * 2);
    unsigned short* atto  = (unsigned short*)alloc((size_t)M * C * 2);

    // 0) zero cls_attn accumulator, write left_tokens
    init_kernel<<<(B * (N - 1) + 255) / 256, 256, 0, stream>>>(
        out_cls, out_left, B * (N - 1));

    // 1) f32 -> bf16 downconversion
    f32_to_bf16_kernel<<<4096, 256, 0, stream>>>(x, xb, M * C);
    f32_to_bf16_kernel<<<2048, 256, 0, stream>>>(qkv_w, wqkv, C * 3 * C);
    f32_to_bf16_kernel<<<1024, 256, 0, stream>>>(proj_w, wproj, C * C);

    // 2) qkv = x @ qkv_w + qkv_b   [12608 x 2304], bf16 out
    gemm_bf16_kernel<true><<<dim3(M / 64, (3 * C) / 256), 256, 0, stream>>>(
        xb, wqkv, qkv_b, qkvo, M, 3 * C, C);

    // 3) attention per (b,h); writes atto in [b,n,h,d] and accumulates cls
    size_t att_lds = (size_t)(2 * ATT_NP * ATT_D + 32 * ATT_D) * 2 +
                     (size_t)32 * ATT_NP * 4;  // 90112 bytes
    attention_kernel<<<dim3(B * H), 256, att_lds, stream>>>(qkvo, out_cls,
                                                            atto);

    // 4) x_out = attn_out @ proj_w + proj_b   [12608 x 768], f32 out
    gemm_bf16_kernel<false><<<dim3(M / 64, C / 256), 256, 0, stream>>>(
        atto, wproj, proj_b, out_x, M, C, C);

    // 5) per-batch top-138 + broadcast index
    topk_kernel<<<dim3(B), 256, 0, stream>>>(out_cls, out_idx, out_index);
}
